// AdaptiveCrossModalAttention_57681410785565
// MI455X (gfx1250) — matrix operand order repaired and checked by reading, not compiled, not run on hardware
//
#include <hip/hip_runtime.h>
#include <hip/hip_bf16.h>
#include <stdint.h>

#define E   768
#define H   16
#define HD  48
#define PD  64
#define BB  4
#define LQ  1024
#define LK  2048
#define SCALE 0.14433756729740643f   // 1/sqrt(48)

typedef __attribute__((ext_vector_type(16))) _Float16 v16h;
typedef __attribute__((ext_vector_type(8)))  float    v8f;

union F16F { v16h h; uint4 u[2]; };

static __device__ __forceinline__ v8f wmma16(v16h a, v16h b, v8f c) {
  // D = A(16x32 f16) * B(32x16 f16) + C(16x16 f32)
  return __builtin_amdgcn_wmma_f32_16x16x32_f16(false, a, false, b, (short)0, c, false, false);
}

// ---------------- workspace layout (bytes) ----------------
constexpr size_t SZ_Q16  = (size_t)BB*H*LQ*PD*2;   //  8 MiB
constexpr size_t SZ_K16  = (size_t)BB*H*LK*PD*2;   // 16 MiB
constexpr size_t SZ_VT   = SZ_K16;                 // 16 MiB
constexpr size_t SZ_CTX  = (size_t)BB*LQ*E*4;      // 12 MiB
constexpr size_t SZ_AO   = SZ_CTX;
constexpr size_t SZ_PIN  = (size_t)BB*2*E*4;
constexpr size_t OFF_Q16 = 0;
constexpr size_t OFF_K16 = OFF_Q16 + SZ_Q16;
constexpr size_t OFF_VT  = OFF_K16 + SZ_K16;
constexpr size_t OFF_CTX = OFF_VT  + SZ_VT;
constexpr size_t OFF_AO  = OFF_CTX + SZ_CTX;
constexpr size_t OFF_PIN = OFF_AO  + SZ_AO;
constexpr size_t OFF_ST  = OFF_PIN + SZ_PIN;
// d_out element offsets
constexpr size_t OUT_AW  = (size_t)BB*LQ*E;            // 3,145,728
constexpr size_t OUT_NH  = OUT_AW + (size_t)BB*LQ*LK;  // 11,534,336

// ---------------- zero fill (pads of f16 staging) ----------------
__global__ void fill_zero_u4(uint4* p, size_t n) {
  size_t i = (size_t)blockIdx.x*256 + threadIdx.x;
  if (i < n) p[i] = make_uint4(0u,0u,0u,0u);
}

// ---------------- pooled means -> pin[B][2E] ----------------
__global__ void mean_kernel(const float* __restrict__ q, const float* __restrict__ k,
                            float* __restrict__ pin) {
  int g = blockIdx.x*256 + threadIdx.x;            // 0..6143
  int b = g / (2*E), c = g % (2*E);
  float s = 0.f;
  if (c < E) {
    const float* p = q + (size_t)b*LQ*E + c;
    for (int l=0;l<LQ;l++) s += p[(size_t)l*E];
    s *= (1.0f/LQ);
  } else {
    const float* p = k + (size_t)b*LK*E + (c-E);
    for (int l=0;l<LK;l++) s += p[(size_t)l*E];
    s *= (1.0f/LK);
  }
  pin[g] = s;
}

// ---------------- tiny MLP head (per batch) ----------------
__global__ void mlp_kernel(const float* __restrict__ pin,
                           const float* __restrict__ Wp1, const float* __restrict__ bp1,
                           const float* __restrict__ Wp2, const float* __restrict__ bp2,
                           const float* __restrict__ Wp3, const float* __restrict__ bp3,
                           float* __restrict__ stats, float* __restrict__ tail_f,
                           int* __restrict__ tail_i) {
  __shared__ float h1[E];
  __shared__ float h2[E/2];
  int b = blockIdx.x, t = threadIdx.x;
  const float* pr = pin + (size_t)b*2*E;
  for (int n=t;n<E;n+=256) {
    float a = bp1[n];
    const float* w = Wp1 + (size_t)n*2*E;
    for (int k2=0;k2<2*E;k2++) a += pr[k2]*w[k2];
    h1[n] = fmaxf(a, 0.f);
  }
  __syncthreads();
  for (int n=t;n<E/2;n+=256) {
    float a = bp2[n];
    const float* w = Wp2 + (size_t)n*E;
    for (int k2=0;k2<E;k2++) a += h1[k2]*w[k2];
    h2[n] = fmaxf(a, 0.f);
  }
  __syncthreads();
  if (t < 3) {
    float a = bp3[t];
    const float* w = Wp3 + (size_t)t*(E/2);
    for (int k2=0;k2<E/2;k2++) a += h2[k2]*w[k2];
    float sig = 1.0f/(1.0f+__expf(-a));
    if (t==0) tail_i[b] = (int)roundf(sig*(H-1)+1.0f);
    if (t==1) { float sf = sig*0.5f+0.5f; stats[b*2+0]=sf; tail_f[BB + b]   = sf; }
    if (t==2) {                            stats[b*2+1]=sig; tail_f[2*BB + b] = sig; }
  }
}

// ---------------- WMMA projection GEMM ----------------
// out[m][n] = sum_k X[m][k]*W[n][k] + bias[n]
// MODE 0/1: f16 out [B][H][L][64] (q16/k16)   MODE 2: f16 out [B][H][64][LK] (vT16)
// MODE 3: f32 out [M][E] (out-projection)
// Each wave owns one 16-row M tile and FOUR 16-col N tiles (A-fragment reuse x4).
template<int MODE, int L>
__global__ void proj_kernel(const float* __restrict__ X, const float* __restrict__ W,
                            const float* __restrict__ bias, _Float16* __restrict__ outh,
                            float* __restrict__ outf) {
  int lane  = threadIdx.x;
  int col   = lane & 15, laneHi = lane >> 4;
  int ng    = blockIdx.y*4 + threadIdx.y;          // 0..11 wave group
  int n0    = ng*64;                               // base column of this wave's 4 N tiles
  int m0    = blockIdx.x*16;
  const float* xr  = X + (size_t)(m0 + col)*E;
  const float* wr0 = W + (size_t)(n0 + col)*E;
  int kbA = laneHi*16;     // A frag: K = kbA+0..15
  int kbB = laneHi*16;     // B frag: K = kbB+0..15
  v8f c0 = {}, c1 = {}, c2 = {}, c3 = {};
  for (int kk=0; kk<E; kk+=32) {
    F16F a;
    const float* xa = xr + kk + kbA;
#pragma unroll
    for (int e=0;e<8;e++) { a.h[e] = (_Float16)xa[e]; a.h[8+e] = (_Float16)xa[8+e]; }
#pragma unroll
    for (int j=0;j<4;j++) {
      F16F bf;
      const float* wb = wr0 + (size_t)j*16*E + kk + kbB;
#pragma unroll
      for (int e=0;e<16;e++) bf.h[e] = (_Float16)wb[e];
      if (j==0)      c0 = wmma16(a.h, bf.h, c0);
      else if (j==1) c1 = wmma16(a.h, bf.h, c1);
      else if (j==2) c2 = wmma16(a.h, bf.h, c2);
      else           c3 = wmma16(a.h, bf.h, c3);
    }
  }
  int bidx = m0 / L;                               // compile-time L -> shift
#pragma unroll
  for (int j=0;j<4;j++) {
    int n = n0 + j*16 + col;
    float bn = bias[n];
    v8f c = (j==0)?c0:(j==1)?c1:(j==2)?c2:c3;
    int hh = n / HD, d = n % HD;
#pragma unroll
    for (int r=0;r<8;r++) {
      int m = m0 + laneHi*8 + r;
      int l = m - bidx*L;
      float v = c[r] + bn;
      if (MODE == 3)      outf[(size_t)m*E + n] = v;
      else if (MODE == 2) outh[(((size_t)bidx*H + hh)*PD + d)*LK + l] = (_Float16)v;
      else                outh[(((size_t)bidx*H + hh)*L  + l)*PD + d] = (_Float16)v;
    }
  }
}

// ---------------- attention: 1 workgroup = (b, 16-row q tile), 16 waves = heads ----------------
__global__ void __launch_bounds__(512) attn_kernel(
    const _Float16* __restrict__ q16, const _Float16* __restrict__ k16,
    const _Float16* __restrict__ vT,  const unsigned char* __restrict__ mask,
    float* __restrict__ ctx, float* __restrict__ attnw) {
  __shared__ float pbuf[H][16][33];     // per-wave p transpose buffer (C layout -> A layout)
  __shared__ float awacc[16][64];       // attn_weights accumulation over heads
  int tid = threadIdx.x;
  int h = tid >> 5, lane = tid & 31;
  int col = lane & 15, laneHi = lane >> 4;
  int b = blockIdx.x >> 6, qt = blockIdx.x & 63;

  for (int i = tid; i < 16*64; i += 512) awacc[i>>6][i&63] = 0.f;
  __syncthreads();

  // q A-fragments (dims 0..31 and 32..63), loaded once
  const _Float16* qrow = q16 + (((size_t)(b*H + h))*LQ + qt*16 + col)*PD;
  F16F aq0, aq1;
  {
    int kb = laneHi*16;
    aq0.u[0] = *(const uint4*)(qrow + kb);
    aq0.u[1] = *(const uint4*)(qrow + kb + 8);
    aq1.u[0] = *(const uint4*)(qrow + 32 + kb);
    aq1.u[1] = *(const uint4*)(qrow + 32 + kb + 8);
  }
  const _Float16* kbase_p = k16 + ((size_t)(b*H + h))*LK*PD;
  const unsigned char* mrow = mask + (size_t)b*LK;
  int kbB = laneHi*16;

  float m8[8], l8[8];
#pragma unroll
  for (int r=0;r<8;r++) { m8[r] = -3.0e38f; l8[r] = 0.f; }

  // ---- pass 1: row max + sumexp, 32 keys per iteration ----
  for (int kt = 0; kt < LK/32; kt++) {
    int key0 = kt*32;
    v8f s0 = {}, s1 = {};
    const _Float16* krow0 = kbase_p + (size_t)(key0 + col)*PD;
    const _Float16* krow1 = krow0 + 16*PD;
    __builtin_prefetch(krow0 + 32*PD, 0, 0);
    F16F bk0, bk1;
    bk0.u[0] = *(const uint4*)(krow0 + kbB);
    bk0.u[1] = *(const uint4*)(krow0 + kbB + 8);
    bk1.u[0] = *(const uint4*)(krow0 + 32 + kbB);
    bk1.u[1] = *(const uint4*)(krow0 + 32 + kbB + 8);
    s0 = wmma16(aq0.h, bk0.h, s0);
    s0 = wmma16(aq1.h, bk1.h, s0);
    bk0.u[0] = *(const uint4*)(krow1 + kbB);
    bk0.u[1] = *(const uint4*)(krow1 + kbB + 8);
    bk1.u[0] = *(const uint4*)(krow1 + 32 + kbB);
    bk1.u[1] = *(const uint4*)(krow1 + 32 + kbB + 8);
    s1 = wmma16(aq0.h, bk0.h, s1);
    s1 = wmma16(aq1.h, bk1.h, s1);
    bool msk0 = mrow[key0 + col]      != 0;
    bool msk1 = mrow[key0 + 16 + col] != 0;
#pragma unroll
    for (int r=0;r<8;r++) {
      float sc0 = msk0 ? -1.0e30f : s0[r]*SCALE;
      float sc1 = msk1 ? -1.0e30f : s1[r]*SCALE;
      float mx = fmaxf(sc0, sc1);
      mx = fmaxf(mx, __shfl_xor(mx, 8, 16));
      mx = fmaxf(mx, __shfl_xor(mx, 4, 16));
      mx = fmaxf(mx, __shfl_xor(mx, 2, 16));
      mx = fmaxf(mx, __shfl_xor(mx, 1, 16));
      float mn = fmaxf(m8[r], mx);
      float ps = __expf(sc0 - mn) + __expf(sc1 - mn);
      ps += __shfl_xor(ps, 8, 16);
      ps += __shfl_xor(ps, 4, 16);
      ps += __shfl_xor(ps, 2, 16);
      ps += __shfl_xor(ps, 1, 16);
      l8[r] = l8[r]*__expf(m8[r]-mn) + ps;
      m8[r] = mn;
    }
  }
  float linv[8];
#pragma unroll
  for (int r=0;r<8;r++) linv[r] = 1.0f / l8[r];

  // ---- pass 2: normalized probs -> attn_weights + ctx ----
  const _Float16* vbase = vT + ((size_t)(b*H + h))*PD*LK;
  v8f ct0 = {}, ct1 = {}, ct2 = {};
  float* aw_base = attnw + ((size_t)b*LQ + qt*16)*LK;

  for (int kc = 0; kc < LK/64; kc++) {
    for (int sub = 0; sub < 2; sub++) {
      int keyb = kc*64 + sub*32;
#pragma unroll
      for (int t2 = 0; t2 < 2; t2++) {
        const _Float16* krow = kbase_p + (size_t)(keyb + t2*16 + col)*PD;
        F16F bk0, bk1;
        bk0.u[0] = *(const uint4*)(krow + kbB);
        bk0.u[1] = *(const uint4*)(krow + kbB + 8);
        bk1.u[0] = *(const uint4*)(krow + 32 + kbB);
        bk1.u[1] = *(const uint4*)(krow + 32 + kbB + 8);
        v8f s = {};
        s = wmma16(aq0.h, bk0.h, s);
        s = wmma16(aq1.h, bk1.h, s);
        bool msk = mrow[keyb + t2*16 + col] != 0;
#pragma unroll
        for (int r=0;r<8;r++) {
          float p = msk ? 0.f : __expf(s[r]*SCALE - m8[r]) * linv[r];
          pbuf[h][laneHi*8 + r][t2*16 + col] = p;
          atomicAdd(&awacc[laneHi*8 + r][sub*32 + t2*16 + col], p * (1.0f/H));
        }
      }
      // p C-layout -> A fragment (wave-private buffer: wave32 lockstep + in-order LDS)
      F16F ap;
      {
        int kb = laneHi*16;
        const float* pr = &pbuf[h][col][0];
#pragma unroll
        for (int e=0;e<8;e++) ap.h[e]   = (_Float16)pr[kb + e];
#pragma unroll
        for (int e=0;e<8;e++) ap.h[8+e] = (_Float16)pr[kb + 8 + e];
      }
#pragma unroll
      for (int t2=0; t2<3; t2++) {     // only 3 real 16-dim tiles (48 of 64)
        const _Float16* vrow = vbase + (size_t)(t2*16 + col)*LK + keyb + kbB;
        F16F bv;
        bv.u[0] = *(const uint4*)(vrow);
        bv.u[1] = *(const uint4*)(vrow + 8);
        if (t2==0)      ct0 = wmma16(ap.h, bv.h, ct0);
        else if (t2==1) ct1 = wmma16(ap.h, bv.h, ct1);
        else            ct2 = wmma16(ap.h, bv.h, ct2);
      }
    }
    __syncthreads();                   // all heads' adds for this 64-key chunk done
    if (tid < 256) {                   // flush as float4 (exclusive rows, no global atomics)
      int rr = tid >> 4, c4 = tid & 15;
      float4* src = (float4*)&awacc[rr][c4*4];
      *(float4*)&aw_base[(size_t)rr*LK + kc*64 + c4*4] = *src;
      *src = make_float4(0.f,0.f,0.f,0.f);
    }
    __syncthreads();
  }
  // ctx store: [B][Lq][E], head h owns dims h*48..h*48+47
  float* crow = ctx + ((size_t)b*LQ + qt*16)*E + h*HD;
#pragma unroll
  for (int r=0;r<8;r++) {
    int m = laneHi*8 + r;
    crow[(size_t)m*E + col]      = ct0[r];
    crow[(size_t)m*E + 16 + col] = ct1[r];
    crow[(size_t)m*E + 32 + col] = ct2[r];
  }
}

// ---------------- gating + residual + LayerNorm ----------------
__global__ void gate_ln_kernel(const float* __restrict__ query, const float* __restrict__ ao,
                               const float* __restrict__ stats, const float* __restrict__ g,
                               const float* __restrict__ bb, float* __restrict__ out) {
  __shared__ float red[256];
  __shared__ float red2[256];
  int rowI = blockIdx.x;
  int b = rowI >> 10;
  int t = threadIdx.x;
  float sf = stats[b*2+0], gv = stats[b*2+1];
  const float* qr = query + (size_t)rowI*E;
  const float* ar = ao    + (size_t)rowI*E;
  float x[3]; float s=0.f, s2=0.f;
#pragma unroll
  for (int j=0;j<3;j++) {
    int e = t + j*256;
    float xv = qr[e]*(2.0f-gv) + ar[e]*sf*gv;   // q + (q*(1-gv) + ao*sf*gv)
    x[j]=xv; s+=xv; s2+=xv*xv;
  }
  red[t]=s; red2[t]=s2; __syncthreads();
  for (int o=128;o>0;o>>=1) { if (t<o){red[t]+=red[t+o];red2[t]+=red2[t+o];} __syncthreads(); }
  float mu  = red[0]*(1.0f/E);
  float var = red2[0]*(1.0f/E) - mu*mu;
  float inv = rsqrtf(var + 1e-5f);
  float* orow = out + (size_t)rowI*E;
#pragma unroll
  for (int j=0;j<3;j++) {
    int e = t + j*256;
    orow[e] = (x[j]-mu)*inv*g[e] + bb[e];
  }
}

extern "C" void kernel_launch(void* const* d_in, const int* in_sizes, int n_in,
                              void* d_out, int out_size, void* d_ws, size_t ws_size,
                              hipStream_t stream) {
  (void)in_sizes; (void)n_in; (void)out_size; (void)ws_size;
  const float* query = (const float*)d_in[0];
  const float* key   = (const float*)d_in[1];
  const float* value = (const float*)d_in[2];
  const unsigned char* mask = (const unsigned char*)d_in[3];
  const float* Wp1 = (const float*)d_in[4];
  const float* bp1 = (const float*)d_in[5];
  const float* Wp2 = (const float*)d_in[6];
  const float* bp2 = (const float*)d_in[7];
  const float* Wp3 = (const float*)d_in[8];
  const float* bp3 = (const float*)d_in[9];
  const float* ipw = (const float*)d_in[10];
  const float* ipb = (const float*)d_in[11];
  const float* outw = (const float*)d_in[12];
  const float* outb = (const float*)d_in[13];
  const float* lng  = (const float*)d_in[14];
  const float* lnb  = (const float*)d_in[15];

  char* ws = (char*)d_ws;
  _Float16* q16  = (_Float16*)(ws + OFF_Q16);
  _Float16* k16  = (_Float16*)(ws + OFF_K16);
  _Float16* vT16 = (_Float16*)(ws + OFF_VT);
  float* ctx   = (float*)(ws + OFF_CTX);
  float* ao    = (float*)(ws + OFF_AO);
  float* pin   = (float*)(ws + OFF_PIN);
  float* stats = (float*)(ws + OFF_ST);

  float* out   = (float*)d_out;
  float* attnw = out + OUT_AW;

  {
    size_t n = (SZ_Q16 + SZ_K16 + SZ_VT)/16;
    fill_zero_u4<<<dim3((unsigned)((n+255)/256)), dim3(256), 0, stream>>>((uint4*)ws, n);
  }
  mean_kernel<<<dim3((BB*2*E)/256), dim3(256), 0, stream>>>(query, key, pin);
  mlp_kernel<<<dim3(BB), dim3(256), 0, stream>>>(pin, Wp1, bp1, Wp2, bp2, Wp3, bp3,
                                                 stats, out + OUT_NH, (int*)d_out + OUT_NH);
  proj_kernel<0,LQ><<<dim3(BB*LQ/16, 3), dim3(32,4), 0, stream>>>(query, ipw,                 ipb,       q16,  nullptr);
  proj_kernel<1,LK><<<dim3(BB*LK/16, 3), dim3(32,4), 0, stream>>>(key,   ipw + (size_t)E*E,   ipb + E,   k16,  nullptr);
  proj_kernel<2,LK><<<dim3(BB*LK/16, 3), dim3(32,4), 0, stream>>>(value, ipw + (size_t)2*E*E, ipb + 2*E, vT16, nullptr);
  attn_kernel<<<dim3(BB*LQ/16), dim3(512), 0, stream>>>(q16, k16, vT16, mask, ctx, attnw);
  proj_kernel<3,LQ><<<dim3(BB*LQ/16, 3), dim3(32,4), 0, stream>>>(ctx, outw, outb, nullptr, ao);
  gate_ln_kernel<<<dim3(BB*LQ), dim3(256), 0, stream>>>(query, ao, stats, lng, lnb, out);
}
